// MFB_35759897706654
// MI455X (gfx1250) — matrix operand order repaired and checked by reading, not compiled, not run on hardware
//
#include <hip/hip_runtime.h>

#define B_   128
#define IN1_ 2048
#define IN2_ 1024
#define F_   256
#define OUT_ 1024

typedef __attribute__((ext_vector_type(16))) _Float16 v16h;
typedef __attribute__((ext_vector_type(8)))  float    v8f;
typedef __attribute__((ext_vector_type(4)))  float    v4f;

// ---------------- projection: h[b,f] = dot(x[b,:], W[f,:]) + bias[f] ----------
// Tiny vs the main GEMM (100 MFLOP, ~5 MB through L2); plain vectorized f32.
__global__ void proj_kernel(const float* __restrict__ x,
                            const float* __restrict__ W,
                            const float* __restrict__ bias,
                            float* __restrict__ h, int K) {
    int idx = blockIdx.x * blockDim.x + threadIdx.x;   // B_*F_ threads
    int b = idx >> 8;          // / F_
    int f = idx & (F_ - 1);    // % F_
    const float* xr = x + (size_t)b * K;
    const float* wr = W + (size_t)f * K;
    v4f acc = {0.f, 0.f, 0.f, 0.f};
    for (int k = 0; k < K; k += 8) {
        v4f xa = *(const v4f*)(xr + k);
        v4f wa = *(const v4f*)(wr + k);
        v4f xb = *(const v4f*)(xr + k + 4);
        v4f wb = *(const v4f*)(wr + k + 4);
        acc += xa * wa + xb * wb;
    }
    h[idx] = acc.x + acc.y + acc.z + acc.w + bias[f];
}

// ---------------- out init: out[b,o] = b_out[o] -------------------------------
__global__ void init_out_kernel(const float* __restrict__ b_out,
                                float* __restrict__ out) {
    int idx = blockIdx.x * blockDim.x + threadIdx.x;   // B_*OUT_ threads
    out[idx] = b_out[idx & (OUT_ - 1)];
}

// ---------------- main GEMM: out += A(h1,h2) x Wout^T -------------------------
// Wave = 16 batch rows x 16 out cols. Workgroup = 8 waves = all 128 batches x
// 16 out cols -> each Wout element crosses HBM exactly once (WGP$ dedupes the
// 8-wave redundancy). Grid.y splits the i-range (K dim) 16 ways; partials are
// combined with fp32 global atomics.
__global__ __launch_bounds__(256) void bilinear_gemm_kernel(
        const float* __restrict__ h1, const float* __restrict__ h2,
        const float* __restrict__ Wout, float* __restrict__ out) {
    const int lane   = threadIdx.x & 31;
    const int wave   = threadIdx.x >> 5;
    const int laneHi = lane >> 4;       // 0 or 1
    const int laneLo = lane & 15;
    const int b0 = wave * 16;           // batch tile base (8 waves -> 128)
    const int n0 = blockIdx.x * 16;     // out-column tile base
    const int i0 = blockIdx.y * 16;     // i-chunk base (ICHUNK = 16)
    const int brow = b0 + laneLo;       // A-layout: lane holds M = lane%16
    const int nrow = n0 + laneLo;       // B-layout: lane holds N = lane%16

    // Preload h1[brow, i0..i0+15] as f16 scalars (A = h1_scalar * h2_vector)
    _Float16 h1h[16];
    {
        const float* p = h1 + (size_t)brow * F_ + i0;
        #pragma unroll
        for (int t = 0; t < 16; t += 4) {
            v4f q = *(const v4f*)(p + t);
            h1h[t + 0] = (_Float16)q.x;  h1h[t + 1] = (_Float16)q.y;
            h1h[t + 2] = (_Float16)q.z;  h1h[t + 3] = (_Float16)q.w;
        }
    }

    v8f acc = {0.f, 0.f, 0.f, 0.f, 0.f, 0.f, 0.f, 0.f};
    const float* wbase = Wout + (size_t)nrow * ((size_t)F_ * F_);

    for (int jb = 0; jb < 8; ++jb) {          // 8 j-blocks of 32 cover F_=256
        const int j0 = jb * 32;
        // Stage h2[brow, j0..j0+31] into A-operand K order:
        //   half m -> K = (m<8 ? m : m+8) + 8*laneHi
        v16h a2;
        {
            const float* p = h2 + (size_t)brow * F_ + j0 + laneHi * 8;
            v4f p0 = *(const v4f*)(p);
            v4f p1 = *(const v4f*)(p + 4);
            v4f p2 = *(const v4f*)(p + 16);
            v4f p3 = *(const v4f*)(p + 20);
            #pragma unroll
            for (int t = 0; t < 4; ++t) {
                a2[t]      = (_Float16)p0[t];
                a2[4 + t]  = (_Float16)p1[t];
                a2[8 + t]  = (_Float16)p2[t];
                a2[12 + t] = (_Float16)p3[t];
            }
        }
        #pragma unroll
        for (int ii = 0; ii < 16; ++ii) {
            // B operand: lane holds N=nrow, halves m -> K = (m<8 ? m : m+8) + 8*laneHi,
            // f32 of Wout row nrow at k = (i0+ii)*F_ + j0
            const float* wp = wbase + (size_t)(i0 + ii) * F_ + j0 + laneHi * 8;
            v4f w0 = *(const v4f*)(wp);
            v4f w1 = *(const v4f*)(wp + 4);
            v4f w2 = *(const v4f*)(wp + 16);
            v4f w3 = *(const v4f*)(wp + 20);
            v16h bop;
            #pragma unroll
            for (int t = 0; t < 4; ++t) {
                bop[t]      = (_Float16)w0[t];
                bop[4 + t]  = (_Float16)w1[t];
                bop[8 + t]  = (_Float16)w2[t];
                bop[12 + t] = (_Float16)w3[t];
            }
            // Rank-1 A tile: outer(h1[:,i], h2[:,j0:j0+32]) -> packed f16 mul
            v16h aop = a2 * h1h[ii];
            acc = __builtin_amdgcn_wmma_f32_16x16x32_f16(
                      false, aop, false, bop, (short)0, acc, false, false);
        }
    }

    // C layout: VGPR r, lanes 0-15 -> M=r, lanes 16-31 -> M=8+r; N = lane%16
    #pragma unroll
    for (int r = 0; r < 8; ++r) {
        int M = r + laneHi * 8;
        atomicAdd(out + (size_t)(b0 + M) * OUT_ + n0 + laneLo, acc[r]);
    }
}

extern "C" void kernel_launch(void* const* d_in, const int* in_sizes, int n_in,
                              void* d_out, int out_size, void* d_ws, size_t ws_size,
                              hipStream_t stream) {
    const float* x1   = (const float*)d_in[0];
    const float* x2   = (const float*)d_in[1];
    const float* W1   = (const float*)d_in[2];
    const float* b1   = (const float*)d_in[3];
    const float* W2   = (const float*)d_in[4];
    const float* b2   = (const float*)d_in[5];
    const float* Wout = (const float*)d_in[6];
    const float* bout = (const float*)d_in[7];
    float* out = (float*)d_out;

    float* h1 = (float*)d_ws;            // 128*256 f32 = 128 KB
    float* h2 = h1 + B_ * F_;            // next 128 KB

    proj_kernel<<<(B_ * F_) / 256, 256, 0, stream>>>(x1, W1, b1, h1, IN1_);
    proj_kernel<<<(B_ * F_) / 256, 256, 0, stream>>>(x2, W2, b2, h2, IN2_);
    init_out_kernel<<<(B_ * OUT_) / 256, 256, 0, stream>>>(bout, out);

    dim3 grid(OUT_ / 16, F_ / 16);       // 64 n-blocks x 16 i-splits
    bilinear_gemm_kernel<<<grid, 256, 0, stream>>>(h1, h2, Wout, out);
}
